// Router_48069273977339
// MI455X (gfx1250) — matrix operand order repaired and checked by reading, not compiled, not run on hardware
//
#include <hip/hip_runtime.h>
#include <cmath>
#include <cstdint>

// ---------------------------------------------------------------------------
// MI455X (gfx1250) fused router: bf16 WMMA GEMMs + TDM-staged LDS score rows
// ---------------------------------------------------------------------------

typedef __attribute__((ext_vector_type(16))) __bf16 v16bf;
typedef __attribute__((ext_vector_type(8)))  __bf16 v8bf;
typedef __attribute__((ext_vector_type(8)))  float  v8f;
typedef __attribute__((ext_vector_type(4)))  unsigned int u32x4;
typedef __attribute__((ext_vector_type(8)))  int i32x8;
typedef __attribute__((ext_vector_type(4)))  int i32x4;

constexpr int B_    = 4;
constexpr int S_    = 2048;
constexpr int D_    = 2048;
constexpr int DB_   = 512;
constexpr int TOKS  = B_ * S_;      // 8192 tokens
constexpr int NQK   = 4096;
constexpr int NV    = 4096;
constexpr int NKNOW = 8192;

// d_out flat offsets (elements), reference tuple order
constexpr size_t OFF_GQ   = 0;
constexpr size_t OFF_IQ   = 524288;   // 4*2048*64
constexpr size_t OFF_GK   = 1048576;
constexpr size_t OFF_IK   = 1572864;
constexpr size_t OFF_GV   = 2097152;
constexpr size_t OFF_IV   = 2621440;
constexpr size_t OFF_AUXA = 3145728;
constexpr size_t OFF_GKN  = 3145729;
constexpr size_t OFF_IKN  = 4194305;  // + 4*2048*128
constexpr size_t OFF_AUXK = 5242881;

__device__ inline v16bf cat8(v8bf lo, v8bf hi) {
  v16bf r;
#pragma unroll
  for (int i = 0; i < 8; ++i) { r[i] = lo[i]; r[i + 8] = hi[i]; }
  return r;
}

// ---------------------------------------------------------------------------
// 1) L2-normalize an embedding table row -> bf16
// ---------------------------------------------------------------------------
__global__ __launch_bounds__(256) void norm_kernel(const float* __restrict__ e,
                                                   __bf16* __restrict__ out) {
  __shared__ float red[256];
  const int row = blockIdx.x, tid = threadIdx.x;
  const float* src = e + (size_t)row * DB_;
  float s = 0.f;
  for (int c = tid; c < DB_; c += 256) { float v = src[c]; s += v * v; }
  red[tid] = s; __syncthreads();
  for (int st = 128; st > 0; st >>= 1) {
    if (tid < st) red[tid] += red[tid + st];
    __syncthreads();
  }
  const float scale = 1.f / (sqrtf(red[0]) + 1e-8f);
  __bf16* dst = out + (size_t)row * DB_;
  for (int c = tid; c < DB_; c += 256) dst[c] = (__bf16)(src[c] * scale);
}

// ---------------------------------------------------------------------------
// 2) H = X @ W + b   (f32 inputs, bf16 output for score stage)
//    block tile 64(M) x 32(N), K-step 32, 8 waves each one 16x16 WMMA acc
// ---------------------------------------------------------------------------
__global__ __launch_bounds__(256) void gemm_xw_bf16_kernel(
    const float* __restrict__ X, const float* __restrict__ W,
    const float* __restrict__ bias, __bf16* __restrict__ H, int ldN) {
  __shared__ __bf16 xLds[64][32];   // [m][k]
  __shared__ __bf16 wLds[32][32];   // [n][k] (transposed stage for B striping)
  const int tid = threadIdx.x;
  const int wv = tid >> 5, lane = tid & 31;
  const int wm = wv & 3, wn = wv >> 2;
  const int laneN = lane & 15, laneH = lane >> 4;
  const int n0 = blockIdx.x * 32, m0 = blockIdx.y * 64;
  const int rS = tid >> 2, cS = (tid & 3) * 8;       // x stage: 64 rows x 32 cols
  const int nS = tid & 31, kS = (tid >> 5) * 4;      // w stage: 32 n x 32 k

  v8f acc = {0.f, 0.f, 0.f, 0.f, 0.f, 0.f, 0.f, 0.f};

  for (int kb = 0; kb < D_; kb += 32) {
    const float* xs = X + (size_t)(m0 + rS) * D_ + kb + cS;
#pragma unroll
    for (int j = 0; j < 8; ++j) xLds[rS][cS + j] = (__bf16)xs[j];
#pragma unroll
    for (int j = 0; j < 4; ++j)
      wLds[nS][kS + j] = (__bf16)W[(size_t)(kb + kS + j) * ldN + n0 + nS];
    __syncthreads();

    // A: lanes 0-15 -> M rows, K-half by lane group (ISA 16-bit A striping)
    v8bf alo = *(const v8bf*)&xLds[wm * 16 + laneN][laneH * 8];
    v8bf ahi = *(const v8bf*)&xLds[wm * 16 + laneN][16 + laneH * 8];
    // B: lane n holds K 0..7,16..23 (lanes 0-15) / 8..15,24..31 (lanes 16-31)
    v8bf blo = *(const v8bf*)&wLds[wn * 16 + laneN][laneH * 8];
    v8bf bhi = *(const v8bf*)&wLds[wn * 16 + laneN][16 + laneH * 8];
    acc = __builtin_amdgcn_wmma_f32_16x16x32_bf16(
        false, cat8(alo, ahi), false, cat8(blo, bhi), (short)0, acc, false, false);
    __syncthreads();
  }

  const int n = n0 + wn * 16 + laneN;
  const float bv = bias[n];
#pragma unroll
  for (int r = 0; r < 8; ++r) {
    const int m = m0 + wm * 16 + r + 8 * laneH;   // C/D layout: VGPR r, lane half
    H[(size_t)m * ldN + n] = (__bf16)(acc[r] + bv);
  }
}

// ---------------------------------------------------------------------------
// 3) tau = x @ [W_tau_attn | W_tau_know] + b   -> tau[tok][4]
// ---------------------------------------------------------------------------
__global__ __launch_bounds__(256) void tau_kernel(
    const float* __restrict__ X, const float* __restrict__ Wa,
    const float* __restrict__ ba, const float* __restrict__ Wk,
    const float* __restrict__ bk, float* __restrict__ tau) {
  __shared__ float red[256];
  const int tg = blockIdx.x, tid = threadIdx.x;
  float a[4] = {0.f, 0.f, 0.f, 0.f};
  for (int d = tid; d < D_; d += 256) {
    const float xv = X[(size_t)tg * D_ + d];
    a[0] += xv * Wa[d * 3 + 0];
    a[1] += xv * Wa[d * 3 + 1];
    a[2] += xv * Wa[d * 3 + 2];
    a[3] += xv * Wk[d];
  }
  for (int c = 0; c < 4; ++c) {
    red[tid] = a[c]; __syncthreads();
    for (int st = 128; st > 0; st >>= 1) {
      if (tid < st) red[tid] += red[tid + st];
      __syncthreads();
    }
    if (tid == 0) tau[(size_t)tg * 4 + c] = red[0] + (c < 3 ? ba[c] : bk[0]);
    __syncthreads();
  }
}

// ---------------------------------------------------------------------------
// 4) Fused scores + softmax-usage + top-k + gate.
//    One workgroup keeps TOK full score rows in LDS (CDNA5: 320KB/WGP).
//    A-operand tile is DMA'd global->LDS by the Tensor Data Mover.
// ---------------------------------------------------------------------------
template <int TOK, int NCOLS, int KSEL, bool USAGE>
__global__ __launch_bounds__(256) void score_gate_kernel(
    const __bf16* __restrict__ h, int hStride,
    const __bf16* __restrict__ embn, const float* __restrict__ tau, int tauCol,
    float* __restrict__ gOut, int* __restrict__ iOut,
    float* __restrict__ usage, float invBS) {
  __shared__ float  sc[TOK][NCOLS];          // 256 KB score rows
  __shared__ __bf16 aLds[16][DB_];           // 16 KB staged A (zero-padded rows)
  __shared__ float  mrow[16], srow[16];
  __shared__ float  topV[8][KSEL];
  __shared__ int    topI[8][KSEL];

  const int tid = threadIdx.x;
  const int wv = tid >> 5, lane = tid & 31;
  const int laneN = lane & 15, laneH = lane >> 4;
  const int tok0 = blockIdx.x * TOK;

#if __has_builtin(__builtin_amdgcn_tensor_load_to_lds)
  // ---- TDM: DMA the TOK x 512 bf16 tile of h into LDS (wave 0 issues) ----
  if (wv == 0) {
    const uint64_t ga  = (uint64_t)(uintptr_t)h + (uint64_t)tok0 * hStride * 2u;
    const uint32_t lds = (uint32_t)(uintptr_t)(void*)&aLds[0][0];  // LDS aperture: addr[31:0]
    u32x4 g0 = {0u, 0u, 0u, 0u};
    g0[0] = 1u;                                         // count=1 (valid user D#)
    g0[1] = lds;                                        // lds_addr (bytes)
    g0[2] = (uint32_t)ga;                               // global_addr[31:0]
    g0[3] = (uint32_t)((ga >> 32) & 0x01FFFFFFu)        // global_addr[56:32]
          | (2u << 30);                                 // type = 2 ("image")
    i32x8 g1 = {0, 0, 0, 0, 0, 0, 0, 0};
    g1[0] = 1 << 16;                                    // data_size = 2 bytes
    g1[1] = (int)(((unsigned)DB_ & 0xFFFFu) << 16);     // tensor_dim0[15:0]
    g1[2] = (int)(((unsigned)TOKS & 0xFFFFu) << 16);    // dim0[31:16]=0 | tensor_dim1[15:0]
    g1[3] = (int)(((unsigned)DB_) << 16);               // dim1[31:16]=0 | tile_dim0 = 512
    g1[4] = TOK;                                        // tile_dim1 = TOK, tile_dim2 = 0
    g1[5] = hStride;                                    // tensor_dim0_stride[31:0]
    g1[6] = 0;                                          // stride0[47:32] | stride1 lo
    g1[7] = 0;
    i32x4 gz = {0, 0, 0, 0};
#if defined(__clang_major__) && (__clang_major__ >= 23)
    __builtin_amdgcn_tensor_load_to_lds(g0, g1, gz, gz, g1, 0);
#else
    __builtin_amdgcn_tensor_load_to_lds(g0, g1, gz, gz, 0);
#endif
    __builtin_amdgcn_s_wait_tensorcnt(0);
  }
  // zero-pad rows [TOK,16) so the M=16 WMMA reads defined data
  for (int idx = TOK * DB_ + tid; idx < 16 * DB_; idx += 256)
    ((__bf16*)aLds)[idx] = (__bf16)0.0f;
#else
  for (int idx = tid; idx < 16 * DB_; idx += 256) {
    const int r = idx >> 9, c = idx & (DB_ - 1);
    __bf16 v = (__bf16)0.0f;
    if (r < TOK) v = h[(size_t)(tok0 + r) * hStride + c];
    aLds[r][c] = v;
  }
#endif
  __syncthreads();

  // ---- scores: each wave owns NCOLS/8 columns, dual 16-col WMMA tiles ----
  const int colsPerWave = NCOLS / 8;
  for (int nc = 0; nc < colsPerWave; nc += 32) {
    const int ncol = wv * colsPerWave + nc;
    const __bf16* b0 = embn + (size_t)(ncol + laneN) * DB_ + laneH * 8;
    const __bf16* b1 = b0 + (size_t)16 * DB_;
    if (nc + 32 < colsPerWave)
      __builtin_prefetch(b0 + (size_t)32 * DB_, 0, 1);   // global_prefetch next pair
    v8f acc0 = {0.f, 0.f, 0.f, 0.f, 0.f, 0.f, 0.f, 0.f};
    v8f acc1 = {0.f, 0.f, 0.f, 0.f, 0.f, 0.f, 0.f, 0.f};
#pragma unroll
    for (int kb = 0; kb < DB_; kb += 32) {
      v16bf a = cat8(*(const v8bf*)&aLds[laneN][kb + laneH * 8],
                     *(const v8bf*)&aLds[laneN][kb + 16 + laneH * 8]);
      v16bf bb0 = cat8(*(const v8bf*)(b0 + kb), *(const v8bf*)(b0 + kb + 16));
      v16bf bb1 = cat8(*(const v8bf*)(b1 + kb), *(const v8bf*)(b1 + kb + 16));
      acc0 = __builtin_amdgcn_wmma_f32_16x16x32_bf16(
          false, a, false, bb0, (short)0, acc0, false, false);
      acc1 = __builtin_amdgcn_wmma_f32_16x16x32_bf16(
          false, a, false, bb1, (short)0, acc1, false, false);
    }
#pragma unroll
    for (int r = 0; r < 8; ++r) {
      const int row = r + 8 * laneH;
      if (row < TOK) {
        sc[row][ncol + laneN]      = acc0[r];
        sc[row][ncol + 16 + laneN] = acc1[r];
      }
    }
  }
  __syncthreads();

  // ---- softmax usage accumulation (only branches the aux loss needs) ----
  if (USAGE) {
    for (int tok = wv; tok < TOK; tok += 8) {
      float m = -__builtin_inff();
      for (int n = lane; n < NCOLS; n += 32) m = fmaxf(m, sc[tok][n]);
#pragma unroll
      for (int off = 16; off > 0; off >>= 1) m = fmaxf(m, __shfl_xor(m, off, 32));
      float s = 0.f;
      for (int n = lane; n < NCOLS; n += 32) s += expf(sc[tok][n] - m);
#pragma unroll
      for (int off = 16; off > 0; off >>= 1) s += __shfl_xor(s, off, 32);
      if (lane == 0) { mrow[tok] = m; srow[tok] = s; }
    }
    __syncthreads();
    for (int n = tid; n < NCOLS; n += 256) {
      float a = 0.f;
#pragma unroll 4
      for (int tok = 0; tok < TOK; ++tok)
        a += expf(sc[tok][n] - mrow[tok]) / srow[tok];
      atomicAdd(&usage[n], a * invBS);
    }
    __syncthreads();
  }

  // ---- iterative wave argmax top-k + threshold gate ----
  for (int tok = wv; tok < TOK; tok += 8) {
    for (int it = 0; it < KSEL; ++it) {
      float best = -__builtin_inff();
      int bidx = 0x7fffffff;
      for (int n = lane; n < NCOLS; n += 32) {
        const float v = sc[tok][n];
        if (v > best || (v == best && n < bidx)) { best = v; bidx = n; }
      }
#pragma unroll
      for (int off = 16; off > 0; off >>= 1) {
        const float ov = __shfl_xor(best, off, 32);
        const int   oi = __shfl_xor(bidx, off, 32);
        if (ov > best || (ov == best && oi < bidx)) { best = ov; bidx = oi; }
      }
      if (lane == 0) {
        topV[wv][it] = best;
        topI[wv][it] = bidx;
        sc[tok][bidx] = -__builtin_inff();
      }
      // wave-level LDS RAW fence before next scan reads lane-0's stores
      asm volatile("s_wait_dscnt 0" ::: "memory");
    }

    const int tg = tok0 + tok;
    const float tv = tau[(size_t)tg * 4 + tauCol];
    float partial = 0.f;
    for (int j = lane; j < KSEL; j += 32) {
      const float raw = topV[wv][j] - tv;
      const float gate = raw > 0.f ? raw : 1e-8f * expf(raw);
      partial += expm1f(gate);
    }
#pragma unroll
    for (int off = 16; off > 0; off >>= 1) partial += __shfl_xor(partial, off, 32);
    const float raw0 = topV[wv][0] - tv;   // top-1 -> max gate (monotone)
    const float gate0 = raw0 > 0.f ? raw0 : 1e-8f * expf(raw0);
    const float strength = tanhf(expm1f(gate0));
    const float denom = partial + 1e-8f;
    for (int j = lane; j < KSEL; j += 32) {
      const float raw = topV[wv][j] - tv;
      const float gate = raw > 0.f ? raw : 1e-8f * expf(raw);
      gOut[(size_t)tg * KSEL + j] = expm1f(gate) / denom * strength;
      iOut[(size_t)tg * KSEL + j] = topI[wv][j];
    }
  }
}

// ---------------------------------------------------------------------------
// 5) zero usage accumulators (graph-replay safe) + aux reduction
// ---------------------------------------------------------------------------
__global__ void zero_kernel(float* __restrict__ p, int n) {
  const int i = blockIdx.x * 256 + threadIdx.x;
  if (i < n) p[i] = 0.f;
}

__global__ __launch_bounds__(256) void aux_kernel(
    const float* __restrict__ uqk, const float* __restrict__ uv,
    const float* __restrict__ ukn, float* __restrict__ auxA,
    float* __restrict__ auxK) {
  __shared__ float red[256];
  const int tid = threadIdx.x;
  const float tq = 1.f / (float)NQK, tk = 1.f / (float)NKNOW;
  float sQ = 0.f, sV = 0.f, sK = 0.f;
  for (int n = tid; n < NQK; n += 256) {
    float d = uqk[n] - tq; sQ += d * d;
    float e = uv[n]  - tq; sV += e * e;
  }
  for (int n = tid; n < NKNOW; n += 256) { float d = ukn[n] - tk; sK += d * d; }

  float tot[3]; float in[3] = {sQ, sV, sK};
  for (int c = 0; c < 3; ++c) {
    red[tid] = in[c]; __syncthreads();
    for (int st = 128; st > 0; st >>= 1) {
      if (tid < st) red[tid] += red[tid + st];
      __syncthreads();
    }
    tot[c] = red[0]; __syncthreads();
  }
  if (tid == 0) {
    *auxA = tot[0] * (float)NQK * 3.f + tot[1] * (float)NV;
    *auxK = tot[2] * (float)NKNOW;
  }
}

// ---------------------------------------------------------------------------
// host launcher
// ---------------------------------------------------------------------------
extern "C" void kernel_launch(void* const* d_in, const int* in_sizes, int n_in,
                              void* d_out, int out_size, void* d_ws,
                              size_t ws_size, hipStream_t stream) {
  (void)in_sizes; (void)n_in; (void)out_size; (void)ws_size;

  const float* x        = (const float*)d_in[0];
  const float* qk_emb   = (const float*)d_in[1];
  const float* v_emb    = (const float*)d_in[2];
  const float* know_emb = (const float*)d_in[3];
  const float* W_attn   = (const float*)d_in[4];
  const float* b_attn   = (const float*)d_in[5];
  const float* W_know   = (const float*)d_in[6];
  const float* b_know   = (const float*)d_in[7];
  const float* W_tau_a  = (const float*)d_in[8];
  const float* b_tau_a  = (const float*)d_in[9];
  const float* W_tau_k  = (const float*)d_in[10];
  const float* b_tau_k  = (const float*)d_in[11];

  char* ws = (char*)d_ws;
  auto carve = [&](size_t bytes) -> char* {
    char* p = ws;
    ws += (bytes + 255) & ~(size_t)255;
    return p;
  };
  __bf16* qk_n   = (__bf16*)carve((size_t)NQK   * DB_ * 2);
  __bf16* v_n    = (__bf16*)carve((size_t)NV    * DB_ * 2);
  __bf16* know_n = (__bf16*)carve((size_t)NKNOW * DB_ * 2);
  __bf16* h_attn = (__bf16*)carve((size_t)TOKS * 1536 * 2);
  __bf16* h_know = (__bf16*)carve((size_t)TOKS * DB_ * 2);
  float*  tau_ws = (float*)carve((size_t)TOKS * 4 * 4);
  float*  usage  = (float*)carve((size_t)(NQK + NV + NKNOW) * 4);
  float* usage_qk = usage;
  float* usage_v  = usage + NQK;
  float* usage_kn = usage + NQK + NV;

  float* out = (float*)d_out;
  const float invBS = 1.f / (float)TOKS;

  // independent prep stages (serialize naturally on one stream)
  zero_kernel<<<(NQK + NV + NKNOW + 255) / 256, 256, 0, stream>>>(usage, NQK + NV + NKNOW);
  norm_kernel<<<NQK,   256, 0, stream>>>(qk_emb,   qk_n);
  norm_kernel<<<NV,    256, 0, stream>>>(v_emb,    v_n);
  norm_kernel<<<NKNOW, 256, 0, stream>>>(know_emb, know_n);
  gemm_xw_bf16_kernel<<<dim3(1536 / 32, TOKS / 64), 256, 0, stream>>>(x, W_attn, b_attn, h_attn, 1536);
  gemm_xw_bf16_kernel<<<dim3(DB_ / 32, TOKS / 64), 256, 0, stream>>>(x, W_know, b_know, h_know, DB_);
  tau_kernel<<<TOKS, 256, 0, stream>>>(x, W_tau_a, b_tau_a, W_tau_k, b_tau_k, tau_ws);

  // fused score + usage + top-k + gate per branch
  score_gate_kernel<16, NQK, 64, true><<<TOKS / 16, 256, 0, stream>>>(
      h_attn, 1536, qk_n, tau_ws, 0, out + OFF_GQ, (int*)(out + OFF_IQ), usage_qk, invBS);
  score_gate_kernel<16, NQK, 64, false><<<TOKS / 16, 256, 0, stream>>>(
      h_attn + 512, 1536, qk_n, tau_ws, 1, out + OFF_GK, (int*)(out + OFF_IK), nullptr, invBS);
  score_gate_kernel<16, NV, 64, true><<<TOKS / 16, 256, 0, stream>>>(
      h_attn + 1024, 1536, v_n, tau_ws, 2, out + OFF_GV, (int*)(out + OFF_IV), usage_v, invBS);
  score_gate_kernel<8, NKNOW, 128, true><<<TOKS / 8, 256, 0, stream>>>(
      h_know, DB_, know_n, tau_ws, 3, out + OFF_GKN, (int*)(out + OFF_IKN), usage_kn, invBS);

  aux_kernel<<<1, 256, 0, stream>>>(usage_qk, usage_v, usage_kn,
                                    out + OFF_AUXA, out + OFF_AUXK);
}
